// Decoder_8143257993354
// MI455X (gfx1250) — matrix operand order repaired and checked by reading, not compiled, not run on hardware
//
#include <hip/hip_runtime.h>
#include <cstdint>
#include <cstddef>

// Problem constants (fixed by the reference)
#define LAYERS 2
#define BATCH  64
#define HID    1024
#define OUTD   1024
#define SEQ    512
#define NKB    (HID / 32)   // 32 k-blocks of 32

typedef unsigned short u16;
typedef __attribute__((ext_vector_type(16))) __bf16          v16bf;
typedef __attribute__((ext_vector_type(16))) unsigned short  v16us;
typedef __attribute__((ext_vector_type(8)))  unsigned short  v8us;
typedef __attribute__((ext_vector_type(8)))  float           v8f;

// ---------------------------------------------------------------------------
// Helpers
// ---------------------------------------------------------------------------
__device__ __forceinline__ u16 f2bf(float f) {
    // round-to-nearest-even fp32 -> bf16
    unsigned int u = __float_as_uint(f);
    unsigned int r = u + 0x7FFFu + ((u >> 16) & 1u);
    return (u16)(r >> 16);
}

__device__ __forceinline__ float sigm(float x) {
    return 1.0f / (1.0f + __expf(-x));
}

// One 16x32 bf16 A-fragment (wave32 WMMA layout) from a bf16 activation row
// pointer (already offset by row*HID + lane K-base).  Each half is 16
// contiguous bytes -> two b128 loads, zero conversion VALU.
__device__ __forceinline__ v16bf load_a2(const u16* __restrict__ ap, int k0) {
    v8us a0 = *reinterpret_cast<const v8us*>(ap + k0);        // K k0..+7
    v8us a1 = *reinterpret_cast<const v8us*>(ap + k0 + 8);    // K k0+8..+15
    v16us u;
#pragma unroll
    for (int j = 0; j < 8; ++j) {
        u[j]     = a0[j];
        u[8 + j] = a1[j];
    }
    return __builtin_bit_cast(v16bf, u);
}

__device__ __forceinline__ v8f wmma_bf16(v16bf a, v16bf b, v8f c) {
    return __builtin_amdgcn_wmma_f32_16x16x32_bf16(
        false, a, false, b, (short)0, c, false, false);
}

// Two-gate GEMM over k-blocks [kb0,kb1), software pipelined with prefetch
// distance 2 (2 stages of 1 A-frag + 2 B-frags in flight over 2 WMMAs).
// Fully unrolled so the rotation copies vanish and the scheduler can keep
// ~12 b128 loads outstanding.
__device__ __forceinline__ void gemm2_pipe(const u16* __restrict__ ap,
                                           const u16* __restrict__ wp0,
                                           const u16* __restrict__ wp1,
                                           int kb0, int kb1,
                                           v8f* accA, v8f* accB) {
    v16bf a0 = load_a2(ap, kb0 * 32);
    v16bf b00 = *reinterpret_cast<const v16bf*>(wp0 + kb0 * 32);
    v16bf b01 = *reinterpret_cast<const v16bf*>(wp1 + kb0 * 32);
    v16bf a1{}, b10{}, b11{};
    if (kb0 + 1 < kb1) {
        a1  = load_a2(ap, (kb0 + 1) * 32);
        b10 = *reinterpret_cast<const v16bf*>(wp0 + (kb0 + 1) * 32);
        b11 = *reinterpret_cast<const v16bf*>(wp1 + (kb0 + 1) * 32);
    }
#pragma unroll
    for (int kb = kb0; kb < kb1; ++kb) {
        v16bf a2{}, b20{}, b21{};
        if (kb + 2 < kb1) {
            a2  = load_a2(ap, (kb + 2) * 32);
            b20 = *reinterpret_cast<const v16bf*>(wp0 + (kb + 2) * 32);
            b21 = *reinterpret_cast<const v16bf*>(wp1 + (kb + 2) * 32);
        }
        *accA = wmma_bf16(a0, b00, *accA);
        *accB = wmma_bf16(a0, b01, *accB);
        a0 = a1; b00 = b10; b01 = b11;
        a1 = a2; b10 = b20; b11 = b21;
    }
}

// Single-tile GEMM over k-blocks [kb0,kb1), prefetch distance 2.
__device__ __forceinline__ void gemm1_pipe(const u16* __restrict__ ap,
                                           const u16* __restrict__ wp,
                                           int kb0, int kb1, v8f* acc) {
    v16bf a0 = load_a2(ap, kb0 * 32);
    v16bf b0 = *reinterpret_cast<const v16bf*>(wp + kb0 * 32);
    v16bf a1{}, b1{};
    if (kb0 + 1 < kb1) {
        a1 = load_a2(ap, (kb0 + 1) * 32);
        b1 = *reinterpret_cast<const v16bf*>(wp + (kb0 + 1) * 32);
    }
#pragma unroll
    for (int kb = kb0; kb < kb1; ++kb) {
        v16bf a2{}, b2{};
        if (kb + 2 < kb1) {
            a2 = load_a2(ap, (kb + 2) * 32);
            b2 = *reinterpret_cast<const v16bf*>(wp + (kb + 2) * 32);
        }
        *acc = wmma_bf16(a0, b0, *acc);
        a0 = a1; b0 = b1;
        a1 = a2; b1 = b2;
    }
}

// ---------------------------------------------------------------------------
// Kernels
// ---------------------------------------------------------------------------

// fp32 -> bf16 conversion (one-time per launch, for weights)
__global__ void cvt_bf16_kernel(const float* __restrict__ src,
                                u16* __restrict__ dst, int n) {
    for (int i = blockIdx.x * blockDim.x + threadIdx.x; i < n;
         i += gridDim.x * blockDim.x)
        dst[i] = f2bf(src[i]);
}

// Seed state: h (bf16) and c (fp32) into parity-1 buffers; broadcast
// init_vec into x0 (bf16).
__global__ void init_state_kernel(const float* __restrict__ hidden,
                                  const float* __restrict__ cell,
                                  const float* __restrict__ initv,
                                  u16* __restrict__ h0, float* __restrict__ c0,
                                  u16* __restrict__ h1, float* __restrict__ c1,
                                  u16* __restrict__ x0) {
    int i = blockIdx.x * blockDim.x + threadIdx.x;
    if (i < BATCH * HID) {
        h0[i] = f2bf(hidden[i]);
        c0[i] = cell[i];
        h1[i] = f2bf(hidden[BATCH * HID + i]);
        c1[i] = cell[BATCH * HID + i];
        x0[i] = f2bf(initv[i & (HID - 1)]);
    }
}

// One LSTM layer for one timestep.  One 128-thread block (4 waves) owns a
// 16x16 h/c tile.  Work split 4 ways, {input GEMM, recurrent GEMM} x
// {gates i,f | gates g,o}, so each wave carries only 2 accumulators and a
// 2-deep load pipeline fits in registers.  Waves 1-3 spill partials through
// LDS; wave 0 combines all four gates and runs the fused LSTM elementwise.
// h state is bf16 (only ever a WMMA A-matrix); c stays fp32.
__global__ __launch_bounds__(128) void
lstm_layer_kernel(const u16* __restrict__ x,       // bf16 [B,H]
                  const u16* __restrict__ hprev,   // bf16 [B,H]
                  const float* __restrict__ cprev, // fp32 [B,H]
                  const u16* __restrict__ Wih,     // bf16 [4H,H]
                  const u16* __restrict__ Whh,     // bf16 [4H,H]
                  const float* __restrict__ bih,
                  const float* __restrict__ bhh,
                  u16* __restrict__ hout,          // bf16 [B,H]
                  float* __restrict__ cout) {      // fp32 [B,H]
    const int lane  = threadIdx.x & 31;
    const int w     = threadIdx.x >> 5;   // 0..3
    const int gpair = w & 1;               // 0: gates {i,f}, 1: gates {g,o}
    const int phase = w >> 1;              // 0: x@Wih^T, 1: h@Whh^T
    const int tile = blockIdx.x;           // 0..255
    const int mt = tile >> 6;              // 4 batch tiles
    const int nt = tile & 63;              // 64 hidden tiles
    const int m0 = mt * 16, n0 = nt * 16;
    const int ln = lane & 15;
    const bool lo = lane < 16;

    __shared__ v8f red[3 * 32 * 2];        // waves 1-3 partials (6 KB)

    const u16* act = phase ? hprev : x;
    const u16* W   = phase ? Whh : Wih;
    const int g0 = gpair * 2;

    const u16* ap  = act + (size_t)(m0 + ln) * HID + (lo ? 0 : 16);
    const u16* wp0 = W + (size_t)((g0 + 0) * HID + n0 + ln) * HID + (lo ? 0 : 16);
    const u16* wp1 = W + (size_t)((g0 + 1) * HID + n0 + ln) * HID + (lo ? 0 : 16);

    v8f accA = {0.f, 0.f, 0.f, 0.f, 0.f, 0.f, 0.f, 0.f};
    v8f accB = accA;
    gemm2_pipe(ap, wp0, wp1, 0, NKB, &accA, &accB);

    if (w != 0) {
        red[((w - 1) * 32 + lane) * 2 + 0] = accA;
        red[((w - 1) * 32 + lane) * 2 + 1] = accB;
    }
    __syncthreads();
    if (w != 0) return;

    // wave layout: w0 = (ih, i/f), w1 = (ih, g/o), w2 = (hh, i/f), w3 = (hh, g/o)
    const v8f i_ = accA + red[(1 * 32 + lane) * 2 + 0];
    const v8f f_ = accB + red[(1 * 32 + lane) * 2 + 1];
    const v8f g_ = red[(0 * 32 + lane) * 2 + 0] + red[(2 * 32 + lane) * 2 + 0];
    const v8f o_ = red[(0 * 32 + lane) * 2 + 1] + red[(2 * 32 + lane) * 2 + 1];

    // Epilogue: C layout -> lane holds column n = n0+ln, rows m0 + r (+8 hi)
    const int n = n0 + ln;
    const float bi = bih[0 * HID + n] + bhh[0 * HID + n];
    const float bf = bih[1 * HID + n] + bhh[1 * HID + n];
    const float bg = bih[2 * HID + n] + bhh[2 * HID + n];
    const float bo = bih[3 * HID + n] + bhh[3 * HID + n];

#pragma unroll
    for (int r = 0; r < 8; ++r) {
        const int m = m0 + r + (lo ? 0 : 8);
        const float iv = i_[r] + bi;
        const float fv = f_[r] + bf;
        const float gv = g_[r] + bg;
        const float ov = o_[r] + bo;
        const float co = cprev[(size_t)m * HID + n];
        const float cn = sigm(fv) * co + sigm(iv) * tanhf(gv);
        const float hn = sigm(ov) * tanhf(cn);
        cout[(size_t)m * HID + n] = cn;
        hout[(size_t)m * HID + n] = f2bf(hn);
    }
}

// FC head: y[b, t, o] = h1[b, :] @ fc_w[o, :]^T + fc_b[o]
// 4-way split-K across the waves of the block, LDS reduce, wave-0 epilogue.
__global__ __launch_bounds__(128) void
fc_kernel(const u16* __restrict__ h,       // bf16 [B,H]
          const u16* __restrict__ W,       // bf16 [OUTD, HID]
          const float* __restrict__ bias,
          float* __restrict__ out, int t) {
    const int lane = threadIdx.x & 31;
    const int w    = threadIdx.x >> 5;     // 0..3
    const int tile = blockIdx.x;           // 0..255
    const int mt = tile >> 6;
    const int nt = tile & 63;
    const int m0 = mt * 16, n0 = nt * 16;
    const int ln = lane & 15;
    const bool lo = lane < 16;

    __shared__ v8f red[3 * 32];            // waves 1-3 partials (3 KB)

    const u16* ap = h + (size_t)(m0 + ln) * HID + (lo ? 0 : 16);
    const u16* wp = W + (size_t)(n0 + ln) * HID + (lo ? 0 : 16);

    v8f acc = {0.f, 0.f, 0.f, 0.f, 0.f, 0.f, 0.f, 0.f};
    const int kq = NKB / 4;                // 8 k-blocks per wave
    gemm1_pipe(ap, wp, w * kq, (w + 1) * kq, &acc);

    if (w != 0)
        red[(w - 1) * 32 + lane] = acc;
    __syncthreads();
    if (w != 0) return;

    acc += red[0 * 32 + lane];
    acc += red[1 * 32 + lane];
    acc += red[2 * 32 + lane];

    const int n = n0 + ln;
    const float bz = bias[n];
#pragma unroll
    for (int r = 0; r < 8; ++r) {
        const int m = m0 + r + (lo ? 0 : 8);
        out[((size_t)m * SEQ + t) * OUTD + n] = acc[r] + bz;
    }
}

// ---------------------------------------------------------------------------
// Host launcher (graph-capture safe: stream-only, deterministic)
// ---------------------------------------------------------------------------
extern "C" void kernel_launch(void* const* d_in, const int* in_sizes, int n_in,
                              void* d_out, int out_size, void* d_ws,
                              size_t ws_size, hipStream_t stream) {
    (void)in_sizes; (void)n_in; (void)out_size; (void)ws_size;

    const float* hidden   = (const float*)d_in[0];  // [L,B,H]
    const float* cell     = (const float*)d_in[1];  // [L,B,H]
    const float* W_ih     = (const float*)d_in[2];  // [L,4H,H]
    const float* W_hh     = (const float*)d_in[3];  // [L,4H,H]
    const float* b_ih     = (const float*)d_in[4];  // [L,4H]
    const float* b_hh     = (const float*)d_in[5];  // [L,4H]
    const float* fc_w     = (const float*)d_in[6];  // [OUT,H]
    const float* fc_b     = (const float*)d_in[7];  // [OUT]
    const float* init_vec = (const float*)d_in[8];  // [1,1,H]
    float* out            = (float*)d_out;          // [B,T,OUT]

    // Workspace carve-up (all offsets keep >=32B alignment)
    char* ws = (char*)d_ws;
    const size_t W_E   = (size_t)LAYERS * 4 * HID * HID;   // 8,388,608
    const size_t FCW_E = (size_t)OUTD * HID;               // 1,048,576
    const size_t ST_E  = (size_t)BATCH * HID;              // 65,536

    u16* wih_bf = (u16*)ws; ws += W_E * sizeof(u16);
    u16* whh_bf = (u16*)ws; ws += W_E * sizeof(u16);
    u16* fcw_bf = (u16*)ws; ws += FCW_E * sizeof(u16);

    u16*   hbuf[2][2];
    float* cbuf[2][2];
    for (int l = 0; l < 2; ++l)
        for (int p = 0; p < 2; ++p) {
            hbuf[l][p] = (u16*)ws; ws += ST_E * sizeof(u16);
        }
    for (int l = 0; l < 2; ++l)
        for (int p = 0; p < 2; ++p) {
            cbuf[l][p] = (float*)ws; ws += ST_E * sizeof(float);
        }
    u16* x0 = (u16*)ws; ws += ST_E * sizeof(u16);

    // One-time weight conversion fp32 -> bf16 (stays in 192MB L2 thereafter)
    cvt_bf16_kernel<<<2048, 256, 0, stream>>>(W_ih, wih_bf, (int)W_E);
    cvt_bf16_kernel<<<2048, 256, 0, stream>>>(W_hh, whh_bf, (int)W_E);
    cvt_bf16_kernel<<<2048, 256, 0, stream>>>(fc_w, fcw_bf, (int)FCW_E);

    // Seed state into parity-1 buffers (step 0 reads parity 1)
    init_state_kernel<<<(BATCH * HID + 255) / 256, 256, 0, stream>>>(
        hidden, cell, init_vec,
        hbuf[0][1], cbuf[0][1], hbuf[1][1], cbuf[1][1], x0);

    const size_t l1w = (size_t)4 * HID * HID;  // per-layer weight stride

    for (int t = 0; t < SEQ; ++t) {
        const int par  = t & 1;
        const int prev = par ^ 1;
        const u16* xin = (t == 0) ? x0 : hbuf[1][prev];

        // Layer 0
        lstm_layer_kernel<<<256, 128, 0, stream>>>(
            xin, hbuf[0][prev], cbuf[0][prev],
            wih_bf, whh_bf, b_ih, b_hh,
            hbuf[0][par], cbuf[0][par]);

        // Layer 1 (input = layer-0 output of this step)
        lstm_layer_kernel<<<256, 128, 0, stream>>>(
            hbuf[0][par], hbuf[1][prev], cbuf[1][prev],
            wih_bf + l1w, whh_bf + l1w, b_ih + 4 * HID, b_hh + 4 * HID,
            hbuf[1][par], cbuf[1][par]);

        // FC head -> out[:, t, :]
        fc_kernel<<<256, 128, 0, stream>>>(hbuf[1][par], fcw_bf, fc_b, out, t);
    }
}
